// AggregateAttention_6459630813903
// MI455X (gfx1250) — matrix operand order repaired and checked by reading, not compiled, not run on hardware
//
#include <hip/hip_runtime.h>

// Sizes from reference
#define Bc 4
#define Ac 2
#define Rc 36
#define Dc 2048
#define Nc 500

typedef __attribute__((ext_vector_type(16))) __bf16 v16bf;
typedef __attribute__((ext_vector_type(8)))  float  v8f;
typedef unsigned int u32x4 __attribute__((ext_vector_type(4)));
typedef int          i32x4 __attribute__((ext_vector_type(4)));
typedef int          i32x8 __attribute__((ext_vector_type(8)));

__device__ __forceinline__ v8f wmma_bf16(v16bf a, v16bf b, v8f c) {
  // (neg_a, A, neg_b, B, c_mod, C, reuse_a, reuse_b)
  return __builtin_amdgcn_wmma_f32_16x16x32_bf16(false, a, false, b, (short)0, c,
                                                 false, false);
}

// Pack two contiguous 8-float runs (ISA 16-bit A-fragment layout: per lane the
// 16 K-values are floats [g*8, g*8+8) and [16+g*8, 16+g*8+8) of the row).
__device__ __forceinline__ void pack_a_frag(const float* __restrict__ row, int g,
                                            v16bf& af) {
  const float4 t0 = *(const float4*)(row + g * 8);
  const float4 t1 = *(const float4*)(row + g * 8 + 4);
  const float4 t2 = *(const float4*)(row + 16 + g * 8);
  const float4 t3 = *(const float4*)(row + 16 + g * 8 + 4);
  af[0] = (__bf16)t0.x;  af[1] = (__bf16)t0.y;
  af[2] = (__bf16)t0.z;  af[3] = (__bf16)t0.w;
  af[4] = (__bf16)t1.x;  af[5] = (__bf16)t1.y;
  af[6] = (__bf16)t1.z;  af[7] = (__bf16)t1.w;
  af[8] = (__bf16)t2.x;  af[9] = (__bf16)t2.y;
  af[10] = (__bf16)t2.z; af[11] = (__bf16)t2.w;
  af[12] = (__bf16)t3.x; af[13] = (__bf16)t3.y;
  af[14] = (__bf16)t3.z; af[15] = (__bf16)t3.w;
}

// ---------------------------------------------------------------------------
// Stage 1: xwx[r][n=b*A+a][d] = x[b,r,:] @ wx[a,r] + wx_bias[a,r,d]
// GEMM per (a,r): M=4 (padded to 16), K=2048, N=2048 (128 per block, 16/wave)
// ---------------------------------------------------------------------------
__global__ __launch_bounds__(256) void k_stage1(const float* __restrict__ x,
                                                const float* __restrict__ wx,
                                                const float* __restrict__ wxb,
                                                float* __restrict__ xwx) {
  const int bid = blockIdx.x;
  const int dt  = bid & 15;
  const int ar  = bid >> 4;
  const int r   = ar % Rc;
  const int a   = ar / Rc;
  const int lane = threadIdx.x & 31;
  const int wv   = threadIdx.x >> 5;
  const int dcol = dt * 128 + wv * 16;

  const float* wxm = wx + (size_t)(a * Rc + r) * Dc * Dc;  // D x D, row-major
  const int m  = lane & 15;        // A-fragment row
  const int g  = lane >> 4;        // half-wave (K group select)
  const int kr = m + (g << 4);     // B-fragment K row 0..31

  v8f c = {0.f, 0.f, 0.f, 0.f, 0.f, 0.f, 0.f, 0.f};
  v16bf af;  // rows 4..15 permanently zero (loop-invariant padding)
#pragma unroll
  for (int i = 0; i < 16; ++i) af[i] = (__bf16)0.0f;
  const float* xr = x + ((size_t)m * Rc + r) * Dc;  // valid only for m < Bc

#pragma unroll 2
  for (int k0 = 0; k0 < Dc; k0 += 32) {
    if (m < Bc) pack_a_frag(xr + k0, g, af);
    // B fragment: lane = d column (dcol+m), 16 K rows of wx in table order
    const float* brow = wxm + (size_t)(k0 + kr) * Dc + dcol;
    v16bf bfr;
#pragma unroll
    for (int s = 0; s < 16; ++s) {
      const int kk = (s < 8) ? (g * 8 + s) : (16 + g * 8 + (s - 8));
      bfr[s] = (__bf16)wxm[(size_t)(k0 + kk) * Dc + dcol + m];
    }
    __builtin_prefetch(brow + (size_t)32 * Dc, 0, 1);
    c = wmma_bf16(af, bfr, c);
  }

  // C/D layout: lanes 0-15 hold rows M=0..7; rows 0..3 are valid (b index)
  if (lane < 16) {
    const int d = dcol + lane;
    const float bias = wxb[(size_t)(a * Rc + r) * Dc + d];
#pragma unroll
    for (int i = 0; i < Bc; ++i) {
      const int n = i * Ac + a;  // row id = b*A + a
      xwx[((size_t)r * 8 + n) * Dc + d] = c[i] + bias;
    }
  }
}

// ---------------------------------------------------------------------------
// Stage 2: z[r][n][d] = sum_e wy[r][d][e] * xwx[r][n][e]   (z = wy @ xwx^T)
// GEMM per r: M = 2048 wy rows (128/block, 16/wave), K = 2048, N = 8 (pad 16)
// ---------------------------------------------------------------------------
__global__ __launch_bounds__(256) void k_stage2(const float* __restrict__ wy,
                                                const float* __restrict__ xwx,
                                                float* __restrict__ z) {
  const int bid = blockIdx.x;
  const int dt  = bid & 15;
  const int r   = bid >> 4;
  const int lane = threadIdx.x & 31;
  const int wv   = threadIdx.x >> 5;
  const int d0   = dt * 128 + wv * 16;

  const float* wyr = wy + (size_t)r * Dc * Dc;
  const float* xr  = xwx + (size_t)r * 8 * Dc;
  const int m  = lane & 15;
  const int g  = lane >> 4;
  const int kr = m + (g << 4);

  v8f c = {0.f, 0.f, 0.f, 0.f, 0.f, 0.f, 0.f, 0.f};
  v16bf bfr;  // columns 8..15 permanently zero padding
#pragma unroll
  for (int j = 0; j < 16; ++j) bfr[j] = (__bf16)0.0f;
  const float* arow0 = wyr + (size_t)(d0 + m) * Dc;

#pragma unroll 2
  for (int k0 = 0; k0 < Dc; k0 += 32) {
    // A fragment: row = wy row (d0+m), row-contiguous 128b loads
    v16bf af;
    pack_a_frag(arow0 + k0, g, af);
    // B fragment: lane = column n = xwx row m (m < 8), 16 K values in table order
    if (m < 8) pack_a_frag(xr + (size_t)m * Dc + k0, g, bfr);
    __builtin_prefetch(arow0 + k0 + 32, 0, 1);
    c = wmma_bf16(af, bfr, c);
  }

  // D layout: lane -> col n = lane&15, row M = 8*g + i  -> d = d0 + 8*g + i
  const int n = lane & 15;
  if (n < 8) {
#pragma unroll
    for (int i = 0; i < 8; ++i) {
      const int d = d0 + (g << 3) + i;
      z[((size_t)r * 8 + n) * Dc + d] = c[i];
    }
  }
}

// ---------------------------------------------------------------------------
// cbias[r*8+n] = xwx[r][n][:] . wy_bias[r][:]
// ---------------------------------------------------------------------------
__global__ __launch_bounds__(256) void k_cbias(const float* __restrict__ xwx,
                                               const float* __restrict__ wyb,
                                               float* __restrict__ cbias) {
  const int row = blockIdx.x;  // r*8+n
  const int r   = row >> 3;
  __shared__ float red[256];
  const float* xr = xwx + (size_t)row * Dc;
  const float* br = wyb + (size_t)r * Dc;
  float s = 0.f;
  for (int d = threadIdx.x; d < Dc; d += 256) s += xr[d] * br[d];
  red[threadIdx.x] = s;
  __syncthreads();
  for (int off = 128; off > 0; off >>= 1) {
    if ((int)threadIdx.x < off) red[threadIdx.x] += red[threadIdx.x + off];
    __syncthreads();
  }
  if (threadIdx.x == 0) cbias[row] = red[0];
}

// ---------------------------------------------------------------------------
// logits[r*8+n][np] = (z[r][n][:] . pool[r][np][:] + cbias) / sqrt(D)
// z (64KB) staged into LDS by the Tensor Data Mover (one wave issues a 1-D
// tile descriptor, waits on TENSORcnt, workgroup barrier publishes LDS).
// ---------------------------------------------------------------------------
__global__ __launch_bounds__(256) void k_logits(const float* __restrict__ pool,
                                                const float* __restrict__ z,
                                                const float* __restrict__ cbias,
                                                float* __restrict__ logit) {
  const int bid = blockIdx.x;
  const int ch  = bid & 31;  // 32 chunks * 16 = 512 >= 500
  const int r   = bid >> 5;
  __shared__ float zsh[8 * Dc];  // 64 KB, at LDS offset 0

  const int lane = threadIdx.x & 31;
  const int wv   = threadIdx.x >> 5;

#if defined(__has_builtin) && __has_builtin(__builtin_amdgcn_tensor_load_to_lds)
  if (wv == 0) {
    // Tensor DMA descriptor (ISA ch.8): 1-D tile, 16384 x f32, LDS addr 0.
    const unsigned long long ga =
        (unsigned long long)(const void*)(z + (size_t)r * 8 * Dc);
    u32x4 g0;
    g0[0] = 1u;                                   // count=1 valid descriptor
    g0[1] = 0u;                                   // lds_addr = 0 (zsh)
    g0[2] = (unsigned)(ga & 0xFFFFFFFFu);         // global_addr[31:0]
    g0[3] = (unsigned)((ga >> 32) & 0x1FFFFFFu)   // global_addr[56:32]
            | (2u << 30);                         // type = 2 ("image")
    i32x8 g1;
    g1[0] = (int)(2u << 16);       // data_size = 4 bytes
    g1[1] = (int)(16384u << 16);   // tensor_dim0[15:0] in bits[79:64? -> 31:16]
    g1[2] = (int)(1u << 16);       // tensor_dim0 hi = 0; tensor_dim1 = 1
    g1[3] = (int)(16384u << 16);   // tile_dim0 = 16384 elements
    g1[4] = 1;                     // tile_dim1 = 1, tile_dim2 = 0
    g1[5] = 16384;                 // tensor_dim0_stride
    g1[6] = 0;
    g1[7] = 0;
    i32x4 gz;
    gz[0] = 0; gz[1] = 0; gz[2] = 0; gz[3] = 0;
#if __clang_major__ >= 23
    i32x8 gz8;
#pragma unroll
    for (int i = 0; i < 8; ++i) gz8[i] = 0;
    __builtin_amdgcn_tensor_load_to_lds(g0, g1, gz, gz, gz8, 0);
#else
    __builtin_amdgcn_tensor_load_to_lds(g0, g1, gz, gz, 0);
#endif
#if __has_builtin(__builtin_amdgcn_s_wait_tensorcnt)
    __builtin_amdgcn_s_wait_tensorcnt(0);
#else
    asm volatile("s_wait_tensorcnt 0x0" ::: "memory");
#endif
  }
  __syncthreads();
#else
  for (int i = threadIdx.x; i < 8 * Dc; i += 256)
    zsh[i] = z[(size_t)r * 8 * Dc + i];
  __syncthreads();
#endif

  for (int e = 0; e < 2; ++e) {
    const int np = ch * 16 + wv * 2 + e;
    if (np >= Nc) continue;
    float acc[8] = {0.f, 0.f, 0.f, 0.f, 0.f, 0.f, 0.f, 0.f};
    const float* pr = pool + ((size_t)r * Nc + np) * Dc;
    for (int d = lane; d < Dc; d += 32) {
      const float pv = pr[d];
#pragma unroll
      for (int n = 0; n < 8; ++n) acc[n] += zsh[n * Dc + d] * pv;
    }
#pragma unroll
    for (int n = 0; n < 8; ++n)
      for (int off = 16; off > 0; off >>= 1) acc[n] += __shfl_xor(acc[n], off);
    if (lane == 0) {
      const float rs = 0.02209708691207961f;  // 1/sqrt(2048)
#pragma unroll
      for (int n = 0; n < 8; ++n)
        logit[(size_t)(r * 8 + n) * Nc + np] = (acc[n] + cbias[r * 8 + n]) * rs;
    }
  }
}

// ---------------------------------------------------------------------------
// In-place softmax over Nc=500 per row
// ---------------------------------------------------------------------------
__global__ __launch_bounds__(256) void k_softmax(float* __restrict__ logit) {
  const int row = blockIdx.x;
  float* p = logit + (size_t)row * Nc;
  __shared__ float red[256];
  const int t = threadIdx.x;
  float v0 = (t < Nc) ? p[t] : -3.4e38f;
  float v1 = (t + 256 < Nc) ? p[t + 256] : -3.4e38f;
  red[t] = fmaxf(v0, v1);
  __syncthreads();
  for (int off = 128; off > 0; off >>= 1) {
    if (t < off) red[t] = fmaxf(red[t], red[t + off]);
    __syncthreads();
  }
  const float mx = red[0];
  __syncthreads();
  v0 = (t < Nc) ? __expf(v0 - mx) : 0.f;
  v1 = (t + 256 < Nc) ? __expf(v1 - mx) : 0.f;
  red[t] = v0 + v1;
  __syncthreads();
  for (int off = 128; off > 0; off >>= 1) {
    if (t < off) red[t] += red[t + off];
    __syncthreads();
  }
  const float inv = 1.f / red[0];
  if (t < Nc) p[t] = v0 * inv;
  if (t + 256 < Nc) p[t + 256] = v1 * inv;
}

// ---------------------------------------------------------------------------
// out[n][r][d] = sum_np P[r*8+n][np] * pool[r][np][d]; block per (r, d-tile)
// ---------------------------------------------------------------------------
__global__ __launch_bounds__(256) void k_out(const float* __restrict__ pool,
                                             const float* __restrict__ p,
                                             float* __restrict__ out) {
  const int bid = blockIdx.x;
  const int dt  = bid & 15;
  const int r   = bid >> 4;
  __shared__ float psh[8 * Nc];
  for (int i = threadIdx.x; i < 8 * Nc; i += 256)
    psh[i] = p[(size_t)r * 8 * Nc + i];
  __syncthreads();

  const int cidx = threadIdx.x & 127;
  const int rq   = threadIdx.x >> 7;  // 0/1 -> rows 0..3 / 4..7
  const int d    = dt * 128 + cidx;
  float a0 = 0.f, a1 = 0.f, a2 = 0.f, a3 = 0.f;
  const float* pl = pool + (size_t)r * Nc * Dc + d;
  for (int np = 0; np < Nc; ++np) {
    const float pv = pl[(size_t)np * Dc];
    a0 += psh[(rq * 4 + 0) * Nc + np] * pv;
    a1 += psh[(rq * 4 + 1) * Nc + np] * pv;
    a2 += psh[(rq * 4 + 2) * Nc + np] * pv;
    a3 += psh[(rq * 4 + 3) * Nc + np] * pv;
  }
  out[((size_t)(rq * 4 + 0) * Rc + r) * Dc + d] = a0;
  out[((size_t)(rq * 4 + 1) * Rc + r) * Dc + d] = a1;
  out[((size_t)(rq * 4 + 2) * Rc + r) * Dc + d] = a2;
  out[((size_t)(rq * 4 + 3) * Rc + r) * Dc + d] = a3;
}

extern "C" void kernel_launch(void* const* d_in, const int* in_sizes, int n_in,
                              void* d_out, int out_size, void* d_ws, size_t ws_size,
                              hipStream_t stream) {
  const float* x    = (const float*)d_in[0];  // (B,R,D)
  const float* pool = (const float*)d_in[1];  // (R,N,D)
  const float* wx   = (const float*)d_in[2];  // (A,R,D,D)
  const float* wxb  = (const float*)d_in[3];  // (A,R,1,D)
  const float* wy   = (const float*)d_in[4];  // (R,D,D)
  const float* wyb  = (const float*)d_in[5];  // (R,1,D)
  float* out = (float*)d_out;                 // (B,A,R,1,D)

  char* ws = (char*)d_ws;
  const size_t XWX = (size_t)Rc * 8 * Dc * sizeof(float);  // 2.36 MB
  float* xwx   = (float*)(ws);
  float* z     = (float*)(ws + XWX);
  float* cb    = (float*)(ws + 2 * XWX);
  float* logit = (float*)(ws + 2 * XWX + 4096);

  k_stage1 <<<Ac * Rc * 16, 256, 0, stream>>>(x, wx, wxb, xwx);
  k_stage2 <<<Rc * 16,      256, 0, stream>>>(wy, xwx, z);
  k_cbias  <<<Rc * 8,       256, 0, stream>>>(xwx, wyb, cb);
  k_logits <<<Rc * 32,      256, 0, stream>>>(pool, z, cb, logit);
  k_softmax<<<Rc * 8,       256, 0, stream>>>(logit);
  k_out    <<<Rc * 16,      256, 0, stream>>>(pool, logit, out);
}
